// Video_Former_2D_4827543241036
// MI455X (gfx1250) — matrix operand order repaired and checked by reading, not compiled, not run on hardware
//
#include <hip/hip_runtime.h>
#include <math.h>

// ---------------------------------------------------------------------------
// MI455X (gfx1250, wave32) implementation of the Perceiver-style Video Former.
// All large GEMMs run through v_wmma_f32_16x16x32_bf16 (f32 accumulate).
// Residual stream / LayerNorm stay in f32; matrix operands are bf16.
// ---------------------------------------------------------------------------

#define BATCH 8
#define TT    16
#define NN    256
#define CDIM  1280
#define DD    1024
#define HH    16
#define DHD   64
#define DEPTH 6
#define QQ    16
#define FFD   4096
#define BT    (BATCH * TT)     // 128
#define NKV   (NN + QQ)        // 272

typedef __attribute__((ext_vector_type(16))) __bf16 bf16x16;
typedef __attribute__((ext_vector_type(8)))  __bf16 bf16x8;
typedef __attribute__((ext_vector_type(4)))  __bf16 bf16x4;
typedef __attribute__((ext_vector_type(8)))  float  f32x8;

enum {
  EPI_STORE_BF16 = 0,  // out_bf16 = acc
  EPI_GELU_BF16  = 1,  // out_bf16 = gelu(acc + bias[n])
  EPI_SCALE_BF16 = 2,  // out_bf16 = acc * scale
  EPI_BIAS_POS_F32 = 3,// out_f32 = acc + bias[n] + pos[t][n], t=(m/256)%16
  EPI_RESID_F32  = 4,  // out_f32 = resid[m][n] + acc
  EPI_RESID_BIAS_F32 = 5 // out_f32 = resid[m][n] + acc + bias[n]
};

__device__ __forceinline__ float gelu_exact(float x) {
  return 0.5f * x * (1.0f + erff(x * 0.70710678118654752f));
}

__device__ __forceinline__ f32x8 zero_f32x8() {
  f32x8 z;
#pragma unroll
  for (int i = 0; i < 8; ++i) z[i] = 0.0f;
  return z;
}

// A-fragment (16x32 bf16): lane l<16 holds row m=l with K {0..7,16..23},
// lanes 16..31 hold row m=l-16 with K {8..15,24..31}.  (ISA 7.12.2)
__device__ __forceinline__ bf16x16 load_a_frag(const __bf16* base, int ldk, int lane) {
  const __bf16* p = base + (size_t)(lane & 15) * ldk + ((lane >> 4) << 3);
  union { bf16x16 v; bf16x8 h[2]; } u;
  u.h[0] = *(const bf16x8*)(p);
  u.h[1] = *(const bf16x8*)(p + 16);
  return u.v;
}

// B-fragment (32x16 bf16): lane l holds column n=l&15, K {0..7,16..23} (l<16)
// or K {8..15,24..31} (l>=16), matching the A-fragment order.
__device__ __forceinline__ bf16x16 load_b_frag(const __bf16* base, int ldk, int lane) {
  const __bf16* p = base + (lane & 15) + (size_t)((lane >> 4) << 3) * ldk;
  bf16x16 v;
#pragma unroll
  for (int j = 0; j < 8; ++j) {
    v[j] = p[(size_t)j * ldk];
    v[8 + j] = p[(size_t)(16 + j) * ldk];
  }
  return v;
}

__device__ __forceinline__ f32x8 wmma_bf16(bf16x16 a, bf16x16 b, f32x8 c) {
  return __builtin_amdgcn_wmma_f32_16x16x32_bf16(false, a, false, b, (short)0, c,
                                                 false, false);
}

// ---------------------------------------------------------------------------
// Generic tiled GEMM: C(MxN) = A(MxK,bf16 rowmajor) * B(KxN,bf16 rowmajor)
// Block tile 64x128, 8 waves of 32x32 (2x2 WMMA accumulators each).
// Requires: M % 64 == 0, N % 128 == 0, K % 32 == 0 (true for all call sites).
// ---------------------------------------------------------------------------
template <int EPI>
__global__ __launch_bounds__(256) void gemm_bf16_wmma(
    const __bf16* __restrict__ A, const __bf16* __restrict__ B,
    const float* __restrict__ bias, const float* __restrict__ extra,
    void* __restrict__ outv, int M, int N, int K, float scale) {
  __shared__ __align__(16) __bf16 As[64][40];    // 32 K + pad8
  __shared__ __align__(16) __bf16 Bs[32][136];   // 128 N + pad8

  const int tid  = threadIdx.x;
  const int lane = tid & 31;
  const int wid  = tid >> 5;
  const int m_base = blockIdx.y * 64;
  const int n_base = blockIdx.x * 128;
  const int wm = wid & 1;        // 2 wave rows (32 each)
  const int wn = wid >> 1;       // 4 wave cols (32 each)

  f32x8 acc00 = zero_f32x8();
  f32x8 acc01 = zero_f32x8();
  f32x8 acc10 = zero_f32x8();
  f32x8 acc11 = zero_f32x8();

  const int arow = tid >> 2, akc = (tid & 3) * 8;      // A tile: 64x32
  const int brow = tid >> 3, bnc = (tid & 7) * 16;     // B tile: 32x128

  for (int k0 = 0; k0 < K; k0 += 32) {
    // stage A (coalesced b128 per thread)
    *(bf16x8*)&As[arow][akc] =
        *(const bf16x8*)&A[(size_t)(m_base + arow) * K + k0 + akc];
    // stage B (2x b128 per thread)
    const __bf16* bsrc = &B[(size_t)(k0 + brow) * N + n_base + bnc];
    *(bf16x8*)&Bs[brow][bnc]     = *(const bf16x8*)(bsrc);
    *(bf16x8*)&Bs[brow][bnc + 8] = *(const bf16x8*)(bsrc + 8);
    if (k0 + 32 < K) {   // emits global_prefetch_b8 for the next tiles
      __builtin_prefetch(&A[(size_t)(m_base + arow) * K + k0 + 32 + akc], 0, 0);
      __builtin_prefetch(&B[(size_t)(k0 + 32 + brow) * N + n_base + bnc], 0, 0);
    }
    __syncthreads();

    bf16x16 a0 = load_a_frag(&As[wm * 32][0], 40, lane);
    bf16x16 a1 = load_a_frag(&As[wm * 32 + 16][0], 40, lane);
    bf16x16 b0 = load_b_frag(&Bs[0][wn * 32], 136, lane);
    bf16x16 b1 = load_b_frag(&Bs[0][wn * 32 + 16], 136, lane);
    acc00 = wmma_bf16(a0, b0, acc00);
    acc01 = wmma_bf16(a0, b1, acc01);
    acc10 = wmma_bf16(a1, b0, acc10);
    acc11 = wmma_bf16(a1, b1, acc11);
    __syncthreads();
  }

  // Epilogue. C layout (ISA 7.12.2): elem r of lane l -> (m = r + 8*(l>=16),
  // n = l&15) within the 16x16 tile.
  auto store_tile = [&](const f32x8& acc, int sm, int sn) {
    const int m0 = m_base + wm * 32 + sm * 16;
    const int n0 = n_base + wn * 32 + sn * 16;
    const int nl = n0 + (lane & 15);
    const int mb = m0 + ((lane >> 4) << 3);
#pragma unroll
    for (int r = 0; r < 8; ++r) {
      const int m = mb + r;
      float v = acc[r];
      if constexpr (EPI == EPI_STORE_BF16) {
        ((__bf16*)outv)[(size_t)m * N + nl] = (__bf16)v;
      } else if constexpr (EPI == EPI_GELU_BF16) {
        ((__bf16*)outv)[(size_t)m * N + nl] = (__bf16)gelu_exact(v + bias[nl]);
      } else if constexpr (EPI == EPI_SCALE_BF16) {
        ((__bf16*)outv)[(size_t)m * N + nl] = (__bf16)(v * scale);
      } else if constexpr (EPI == EPI_BIAS_POS_F32) {
        const int t = (m >> 8) & 15;  // row = (b*T + t)*N + n_feat, N=256
        ((float*)outv)[(size_t)m * N + nl] = v + bias[nl] + extra[t * DD + nl];
      } else if constexpr (EPI == EPI_RESID_F32) {
        ((float*)outv)[(size_t)m * N + nl] = extra[(size_t)m * N + nl] + v;
      } else {  // EPI_RESID_BIAS_F32
        ((float*)outv)[(size_t)m * N + nl] =
            extra[(size_t)m * N + nl] + v + bias[nl];
      }
    }
  };
  store_tile(acc00, 0, 0);
  store_tile(acc01, 0, 1);
  store_tile(acc10, 1, 0);
  store_tile(acc11, 1, 1);
}

// ---------------------------------------------------------------------------
// Fused cross-attention for one (bt, head): sim = Q*K^T, softmax, O = P*V.
// Q:16x64  K,V:272x64 (head slice of 1024-wide buffers).  All WMMA.
// Accumulators are named registers (no runtime-indexed arrays -> no scratch).
// ---------------------------------------------------------------------------
__global__ __launch_bounds__(256) void attention_kernel(
    const __bf16* __restrict__ q, const __bf16* __restrict__ k,
    const __bf16* __restrict__ v, __bf16* __restrict__ o) {
  const int tid  = threadIdx.x;
  const int lane = tid & 31;
  const int wid  = tid >> 5;
  const int h    = blockIdx.x & 15;
  const int bt   = blockIdx.x >> 4;

  // Overlay: KT (phase 1) and Sim (phase 2) share storage.
  __shared__ __align__(16) unsigned char smem[2304 + 64 * 280 * 2 + 16 * 296 * 2];
  __bf16 (*Qs)[72]   = (__bf16 (*)[72])(smem);                       // 16x(64+8)
  __bf16 (*KT)[280]  = (__bf16 (*)[280])(smem + 2304);               // 64x(272+8)
  float  (*Sim)[276] = (float  (*)[276])(smem + 2304);               // 16x(272+4)
  __bf16 (*Pm)[296]  = (__bf16 (*)[296])(smem + 2304 + 64 * 280 * 2);// 16x(288+8)

  // ---- stage Q (16x64) ----
  {
    const int row = tid >> 4, c = (tid & 15) * 4;
    *(bf16x4*)&Qs[row][c] =
        *(const bf16x4*)&q[((size_t)bt * QQ + row) * DD + h * DHD + c];
  }
  // ---- stage K transposed: KT[dh][n] = K[n][dh] ----
  for (int it = 0; it < 9; ++it) {
    const int c = tid + it * 256;
    if (c < NKV * 8) {
      const int n = c >> 3, kc = (c & 7) * 8;
      bf16x8 val = *(const bf16x8*)&k[((size_t)bt * NKV + n) * DD + h * DHD + kc];
#pragma unroll
      for (int j = 0; j < 8; ++j) KT[kc + j][n] = val[j];
    }
  }
  __syncthreads();

  // ---- sim = Q * K^T  (M=16, N=272 in 17 tiles, K=64) ----
  // Wave w owns tiles w and w+8; tile 16 is computed by ALL waves (keeps EXEC
  // all-ones, costs 2 redundant WMMAs) but stored only by wave 0.
  const bf16x16 af0 = load_a_frag(&Qs[0][0], 72, lane);
  const bf16x16 af1 = load_a_frag(&Qs[0][32], 72, lane);
  const int t0 = wid, t1 = wid + 8;

  f32x8 s0 = zero_f32x8();
  s0 = wmma_bf16(af0, load_b_frag(&KT[0][t0 * 16], 280, lane), s0);
  s0 = wmma_bf16(af1, load_b_frag(&KT[32][t0 * 16], 280, lane), s0);
  f32x8 s1 = zero_f32x8();
  s1 = wmma_bf16(af0, load_b_frag(&KT[0][t1 * 16], 280, lane), s1);
  s1 = wmma_bf16(af1, load_b_frag(&KT[32][t1 * 16], 280, lane), s1);
  f32x8 s2 = zero_f32x8();
  s2 = wmma_bf16(af0, load_b_frag(&KT[0][16 * 16], 280, lane), s2);
  s2 = wmma_bf16(af1, load_b_frag(&KT[32][16 * 16], 280, lane), s2);

  __syncthreads();  // everyone done reading KT before Sim overwrites it

  {
    const int nl0 = t0 * 16 + (lane & 15);
    const int nl1 = t1 * 16 + (lane & 15);
    const int mb  = (lane >> 4) * 8;
#pragma unroll
    for (int r = 0; r < 8; ++r) Sim[mb + r][nl0] = s0[r];
#pragma unroll
    for (int r = 0; r < 8; ++r) Sim[mb + r][nl1] = s1[r];
    if (wid == 0) {
      const int nl2 = 256 + (lane & 15);
#pragma unroll
      for (int r = 0; r < 8; ++r) Sim[mb + r][nl2] = s2[r];
    }
  }
  __syncthreads();

  // ---- softmax over 272 keys, write P (bf16, zero-padded K to 288) ----
  {
    const int row = tid >> 4, sub = tid & 15;  // 16 threads per row (half-wave)
    float mx = -3.0e38f;
#pragma unroll
    for (int j = 0; j < 17; ++j) mx = fmaxf(mx, Sim[row][sub + j * 16]);
#pragma unroll
    for (int msk = 8; msk; msk >>= 1) mx = fmaxf(mx, __shfl_xor(mx, msk, 16));
    float sum = 0.0f;
#pragma unroll
    for (int j = 0; j < 17; ++j) {
      const int n = sub + j * 16;
      const float e = expf(Sim[row][n] - mx);
      Sim[row][n] = e;
      sum += e;
    }
#pragma unroll
    for (int msk = 8; msk; msk >>= 1) sum += __shfl_xor(sum, msk, 16);
    const float inv = 1.0f / sum;
#pragma unroll
    for (int j = 0; j < 17; ++j) {
      const int n = sub + j * 16;
      Pm[row][n] = (__bf16)(Sim[row][n] * inv);
    }
    Pm[row][272 + sub] = (__bf16)0.0f;  // zero pad K 272..287
  }
  __syncthreads();

  // ---- O = P * V  (M=16, N=64 in 4 tiles, K=288). All 8 waves compute
  // (waves 4..7 duplicate 0..3 so EXEC stays all-ones); only 0..3 store.
  {
    const int n0 = (wid & 3) * 16;
    f32x8 a = zero_f32x8();
#pragma unroll
    for (int k0 = 0; k0 < 288; k0 += 32) {
      bf16x16 af = load_a_frag(&Pm[0][k0], 296, lane);
      bf16x16 bv;
      const int nn = n0 + (lane & 15);
#pragma unroll
      for (int j = 0; j < 16; ++j) {
        const int kk = k0 + ((lane >> 4) << 3) + (j < 8 ? j : j + 8);
        bv[j] = (kk < NKV) ? v[((size_t)bt * NKV + kk) * DD + h * DHD + nn]
                           : (__bf16)0.0f;
      }
      a = wmma_bf16(af, bv, a);
    }
    if (wid < 4) {
      const int nl = n0 + (lane & 15);
      const int mb = (lane >> 4) * 8;
#pragma unroll
      for (int r = 0; r < 8; ++r)
        o[((size_t)bt * QQ + mb + r) * DD + h * DHD + nl] = (__bf16)a[r];
    }
  }
}

// ---------------------------------------------------------------------------
// LayerNorm, f32 in -> bf16 out with optional row remap + second destination.
// dst0 row = (row/seg)*segstride + segoff + (row%seg).
// ---------------------------------------------------------------------------
__global__ __launch_bounds__(256) void ln_to_bf16(
    const float* __restrict__ x, const float* __restrict__ g,
    const float* __restrict__ b, __bf16* __restrict__ out0,
    __bf16* __restrict__ out1, int D, int seg, int segstride, int segoff) {
  const int row = blockIdx.x, tid = threadIdx.x;
  const float* xr = x + (size_t)row * D;
  float s = 0.0f, ss = 0.0f;
  for (int i = tid; i < D; i += 256) {
    const float vv = xr[i];
    s += vv;
    ss += vv * vv;
  }
#pragma unroll
  for (int m = 16; m; m >>= 1) {
    s += __shfl_xor(s, m, 32);
    ss += __shfl_xor(ss, m, 32);
  }
  __shared__ float red[16];
  if ((tid & 31) == 0) {
    red[tid >> 5] = s;
    red[8 + (tid >> 5)] = ss;
  }
  __syncthreads();
  s = 0.0f; ss = 0.0f;
#pragma unroll
  for (int w = 0; w < 8; ++w) { s += red[w]; ss += red[8 + w]; }
  const float mean = s / (float)D;
  const float var = ss / (float)D - mean * mean;
  const float rstd = rsqrtf(var + 1e-5f);
  const size_t r0 = (size_t)(row / seg) * segstride + segoff + (row % seg);
  __bf16* d0 = out0 + r0 * D;
  __bf16* d1 = out1 ? out1 + (size_t)row * D : nullptr;
  for (int i = tid; i < D; i += 256) {
    const float y = (xr[i] - mean) * rstd * g[i] + b[i];
    d0[i] = (__bf16)y;
    if (d1) d1[i] = (__bf16)y;
  }
}

__global__ __launch_bounds__(256) void ln_to_f32(
    const float* __restrict__ x, const float* __restrict__ g,
    const float* __restrict__ b, float* __restrict__ out, int D) {
  const int row = blockIdx.x, tid = threadIdx.x;
  const float* xr = x + (size_t)row * D;
  float s = 0.0f, ss = 0.0f;
  for (int i = tid; i < D; i += 256) {
    const float vv = xr[i];
    s += vv;
    ss += vv * vv;
  }
#pragma unroll
  for (int m = 16; m; m >>= 1) {
    s += __shfl_xor(s, m, 32);
    ss += __shfl_xor(ss, m, 32);
  }
  __shared__ float red[16];
  if ((tid & 31) == 0) {
    red[tid >> 5] = s;
    red[8 + (tid >> 5)] = ss;
  }
  __syncthreads();
  s = 0.0f; ss = 0.0f;
#pragma unroll
  for (int w = 0; w < 8; ++w) { s += red[w]; ss += red[8 + w]; }
  const float mean = s / (float)D;
  const float var = ss / (float)D - mean * mean;
  const float rstd = rsqrtf(var + 1e-5f);
  float* d0 = out + (size_t)row * D;
  for (int i = tid; i < D; i += 256)
    d0[i] = (xr[i] - mean) * rstd * g[i] + b[i];
}

// x[bt][q][d] = latents[bt % T][q][d]
__global__ __launch_bounds__(256) void init_latents(
    const float* __restrict__ latents, float* __restrict__ x, int total) {
  const int i = blockIdx.x * 256 + threadIdx.x;
  if (i < total) {
    const int d = i & 1023;
    const int rowq = i >> 10;
    const int qq = rowq & 15;
    const int bt = rowq >> 4;
    const int t = bt & 15;
    x[i] = latents[(((t << 4) + qq) << 10) + d];
  }
}

__global__ __launch_bounds__(256) void cvt_f32_bf16(
    const float* __restrict__ in, __bf16* __restrict__ out, long n) {
  long i = (long)blockIdx.x * 256 + threadIdx.x;
  const long stride = (long)gridDim.x * 256;
  for (; i < n; i += stride) out[i] = (__bf16)in[i];
}

// ---------------------------------------------------------------------------
// Host orchestration.
// ---------------------------------------------------------------------------
extern "C" void kernel_launch(void* const* d_in, const int* in_sizes, int n_in,
                              void* d_out, int out_size, void* d_ws,
                              size_t ws_size, hipStream_t stream) {
  (void)in_sizes; (void)n_in; (void)out_size; (void)ws_size;

  const float* x_f       = (const float*)d_in[0];
  const float* goal_w1   = (const float*)d_in[1];
  const float* goal_b1   = (const float*)d_in[2];
  const float* goal_w2   = (const float*)d_in[3];
  const float* goal_b2   = (const float*)d_in[4];
  const float* latents   = (const float*)d_in[5];
  const float* time_pos  = (const float*)d_in[6];
  const float* ln_m_g    = (const float*)d_in[7];
  const float* ln_m_b    = (const float*)d_in[8];
  const float* ln_l_g    = (const float*)d_in[9];
  const float* ln_l_b    = (const float*)d_in[10];
  const float* wq        = (const float*)d_in[11];
  const float* wk        = (const float*)d_in[12];
  const float* wv        = (const float*)d_in[13];
  const float* wo        = (const float*)d_in[14];
  const float* ffn_ln_g  = (const float*)d_in[15];
  const float* ffn_ln_b  = (const float*)d_in[16];
  const float* ffn_w1    = (const float*)d_in[17];
  const float* ffn_b1    = (const float*)d_in[18];
  const float* ffn_w2    = (const float*)d_in[19];
  const float* ffn_b2    = (const float*)d_in[20];
  const float* final_g   = (const float*)d_in[21];
  const float* final_b   = (const float*)d_in[22];

  char* ws = (char*)d_ws;
  size_t off = 0;
  auto alloc = [&](size_t bytes) -> char* {
    off = (off + 255) & ~(size_t)255;
    char* p = ws + off;
    off += bytes;
    return p;
  };

  const size_t NXF  = (size_t)BT * NN * CDIM;          // 41,943,040
  const size_t NW1  = (size_t)CDIM * 2 * DD;           // 2,621,440
  const size_t NW2  = (size_t)2 * DD * DD;             // 2,097,152
  const size_t NWP  = (size_t)DEPTH * DD * DD;         // 6,291,456
  const size_t NF1  = (size_t)DEPTH * DD * FFD;        // 25,165,824
  const size_t NH1  = (size_t)BT * NN * 2 * DD;        // 67,108,864
  const size_t NFEAT= (size_t)BT * NN * DD;            // 33,554,432
  const size_t NX   = (size_t)BT * QQ * DD;            // 2,097,152
  const size_t NKVE = (size_t)BT * NKV * DD;           // 35,651,584

  __bf16* xf_bf  = (__bf16*)alloc(NXF * 2);
  __bf16* gw1_bf = (__bf16*)alloc(NW1 * 2);
  __bf16* gw2_bf = (__bf16*)alloc(NW2 * 2);
  __bf16* wq_bf  = (__bf16*)alloc(NWP * 2);
  __bf16* wk_bf  = (__bf16*)alloc(NWP * 2);
  __bf16* wv_bf  = (__bf16*)alloc(NWP * 2);
  __bf16* wo_bf  = (__bf16*)alloc(NWP * 2);
  __bf16* fw1_bf = (__bf16*)alloc(NF1 * 2);
  __bf16* fw2_bf = (__bf16*)alloc(NF1 * 2);
  __bf16* h1_bf  = (__bf16*)alloc(NH1 * 2);
  float*  feats  = (float*) alloc(NFEAT * 4);
  float*  xlat   = (float*) alloc(NX * 4);
  __bf16* kvn_bf = (__bf16*)alloc(NKVE * 2);
  __bf16* latn_bf= (__bf16*)alloc(NX * 2);
  __bf16* q_bf   = (__bf16*)alloc(NX * 2);
  __bf16* k_bf   = (__bf16*)alloc(NKVE * 2);
  __bf16* v_bf   = (__bf16*)alloc(NKVE * 2);
  __bf16* o_bf   = (__bf16*)alloc(NX * 2);
  __bf16* y_bf   = (__bf16*)alloc(NX * 2);
  __bf16* ffh_bf = (__bf16*)alloc((size_t)BT * QQ * FFD * 2);
  // total ~770 MB of workspace

  const int CV = 8192;
  cvt_f32_bf16<<<CV, 256, 0, stream>>>(x_f,     xf_bf,  (long)NXF);
  cvt_f32_bf16<<<CV, 256, 0, stream>>>(goal_w1, gw1_bf, (long)NW1);
  cvt_f32_bf16<<<CV, 256, 0, stream>>>(goal_w2, gw2_bf, (long)NW2);
  cvt_f32_bf16<<<CV, 256, 0, stream>>>(wq,      wq_bf,  (long)NWP);
  cvt_f32_bf16<<<CV, 256, 0, stream>>>(wk,      wk_bf,  (long)NWP);
  cvt_f32_bf16<<<CV, 256, 0, stream>>>(wv,      wv_bf,  (long)NWP);
  cvt_f32_bf16<<<CV, 256, 0, stream>>>(wo,      wo_bf,  (long)NWP);
  cvt_f32_bf16<<<CV, 256, 0, stream>>>(ffn_w1,  fw1_bf, (long)NF1);
  cvt_f32_bf16<<<CV, 256, 0, stream>>>(ffn_w2,  fw2_bf, (long)NF1);

  const int Mfeat = BT * NN;    // 32768
  const int Mlat  = BT * QQ;    // 2048
  const int Mkv   = BT * NKV;   // 34816

  // goal embedding: h1 = GELU(x_f @ w1 + b1)   (32768 x 2048, K=1280)
  gemm_bf16_wmma<EPI_GELU_BF16><<<dim3(2 * DD / 128, Mfeat / 64), 256, 0, stream>>>(
      xf_bf, gw1_bf, goal_b1, nullptr, h1_bf, Mfeat, 2 * DD, CDIM, 1.0f);
  // feats = h1 @ w2 + b2 + time_pos_emb       (32768 x 1024, K=2048)
  gemm_bf16_wmma<EPI_BIAS_POS_F32><<<dim3(DD / 128, Mfeat / 64), 256, 0, stream>>>(
      h1_bf, gw2_bf, goal_b2, time_pos, feats, Mfeat, DD, 2 * DD, 1.0f);
  // x = broadcast latents
  init_latents<<<(int)((NX + 255) / 256), 256, 0, stream>>>(latents, xlat, (int)NX);

  for (int i = 0; i < DEPTH; ++i) {
    const __bf16* wq_i = wq_bf + (size_t)i * DD * DD;
    const __bf16* wk_i = wk_bf + (size_t)i * DD * DD;
    const __bf16* wv_i = wv_bf + (size_t)i * DD * DD;
    const __bf16* wo_i = wo_bf + (size_t)i * DD * DD;
    const __bf16* f1_i = fw1_bf + (size_t)i * DD * FFD;
    const __bf16* f2_i = fw2_bf + (size_t)i * FFD * DD;

    // LN(feats) -> kv rows [0..N); LN(x) -> kv rows [N..N+Q) and latn
    ln_to_bf16<<<Mfeat, 256, 0, stream>>>(feats, ln_m_g + i * DD, ln_m_b + i * DD,
                                          kvn_bf, nullptr, DD, NN, NKV, 0);
    ln_to_bf16<<<Mlat, 256, 0, stream>>>(xlat, ln_l_g + i * DD, ln_l_b + i * DD,
                                         kvn_bf, latn_bf, DD, QQ, NKV, NN);

    // q = (latn @ wq) * DH^-0.5 ; k/v = kvn @ wk/wv
    gemm_bf16_wmma<EPI_SCALE_BF16><<<dim3(DD / 128, Mlat / 64), 256, 0, stream>>>(
        latn_bf, wq_i, nullptr, nullptr, q_bf, Mlat, DD, DD, 0.125f);
    gemm_bf16_wmma<EPI_STORE_BF16><<<dim3(DD / 128, Mkv / 64), 256, 0, stream>>>(
        kvn_bf, wk_i, nullptr, nullptr, k_bf, Mkv, DD, DD, 1.0f);
    gemm_bf16_wmma<EPI_STORE_BF16><<<dim3(DD / 128, Mkv / 64), 256, 0, stream>>>(
        kvn_bf, wv_i, nullptr, nullptr, v_bf, Mkv, DD, DD, 1.0f);

    // fused attention per (bt, head)
    attention_kernel<<<BT * HH, 256, 0, stream>>>(q_bf, k_bf, v_bf, o_bf);

    // x += o @ wo
    gemm_bf16_wmma<EPI_RESID_F32><<<dim3(DD / 128, Mlat / 64), 256, 0, stream>>>(
        o_bf, wo_i, nullptr, xlat, xlat, Mlat, DD, DD, 1.0f);

    // FFN: y = LN(x); x += GELU(y@w1+b1) @ w2 + b2
    ln_to_bf16<<<Mlat, 256, 0, stream>>>(xlat, ffn_ln_g + i * DD, ffn_ln_b + i * DD,
                                         y_bf, nullptr, DD, 1 << 30, 0, 0);
    gemm_bf16_wmma<EPI_GELU_BF16><<<dim3(FFD / 128, Mlat / 64), 256, 0, stream>>>(
        y_bf, f1_i, ffn_b1 + (size_t)i * FFD, nullptr, ffh_bf, Mlat, FFD, DD, 1.0f);
    gemm_bf16_wmma<EPI_RESID_BIAS_F32><<<dim3(DD / 128, Mlat / 64), 256, 0, stream>>>(
        ffh_bf, f2_i, ffn_b2 + (size_t)i * DD, xlat, xlat, Mlat, DD, FFD, 1.0f);
  }

  // final layer norm -> d_out (f32, (B, T*Q, D) == flat (BT*Q, D))
  ln_to_f32<<<Mlat, 256, 0, stream>>>(xlat, final_g, final_b, (float*)d_out, DD);
}
